// GraphSAGEModel_36567351558183
// MI455X (gfx1250) — matrix operand order repaired and checked by reading, not compiled, not run on hardware
//
#include <hip/hip_runtime.h>
#include <math.h>

// ---------------------------------------------------------------------------
// GraphSAGE forward (3x SAGEConv + log_softmax) for MI455X (gfx1250, wave32).
//
// N=50000 (divisible by 16), E=800000, DIN=DH=128, DOUT=8.
//
// Roofline: feature matrices (25.6 MB) fit in 192 MB L2, so the edge scatter
// (~410 MB/layer of gather + atomic-add) runs out of L2; HBM only streams
// ~100 MB total (~4-5 us at 23.3 TB/s). GEMMs (~6.8 GFLOP) ride on
// v_wmma_f32_16x16x32_f16 with f16 A/B and f32 accumulation. Weights are
// staged per-workgroup into LDS via global_load_async_to_lds_b128 (ASYNCcnt)
// so the WMMA inner loop feeds B from ds_load_b128 instead of re-reading
// 64 KB/tile from the vector-memory path.
// ---------------------------------------------------------------------------

typedef __attribute__((ext_vector_type(16))) _Float16 v16h;
typedef __attribute__((ext_vector_type(8)))  float    v8f;

#define NNODES 50000
#define NEDGES 800000
#define DIN    128
#define DH     128
#define DOUT   8

// ---------------------------------------------------------------------------
// small utility kernels
// ---------------------------------------------------------------------------

__global__ void fill_f32(float* p, float v, long n) {
    long i = (long)blockIdx.x * blockDim.x + threadIdx.x;
    long stride = (long)gridDim.x * blockDim.x;
    for (; i < n; i += stride) p[i] = v;
}

__global__ void fill_i32(int* p, int v, long n) {
    long i = (long)blockIdx.x * blockDim.x + threadIdx.x;
    long stride = (long)gridDim.x * blockDim.x;
    for (; i < n; i += stride) p[i] = v;
}

__global__ void f32_to_f16(const float* __restrict__ s, _Float16* __restrict__ d, int n) {
    int i = blockIdx.x * blockDim.x + threadIdx.x;
    if (i < n) d[i] = (_Float16)s[i];
}

// in-degree histogram (dst side), int atomics
__global__ void degree_kernel(const int* __restrict__ ei, int* __restrict__ cnt, int e) {
    int i = blockIdx.x * blockDim.x + threadIdx.x;
    if (i < e) atomicAdd(&cnt[ei[NEDGES + i]], 1);
}

__global__ void deg_inv_kernel(const int* __restrict__ cnt, float* __restrict__ inv, int n) {
    int i = blockIdx.x * blockDim.x + threadIdx.x;
    if (i < n) {
        int c = cnt[i];
        inv[i] = 1.0f / (float)(c > 0 ? c : 1);
    }
}

// ---------------------------------------------------------------------------
// scatter-add: one wave per edge, 4 floats per lane (128 features).
// feat[src] -> agg[dst]. L2-resident global_atomic_add_f32.
// ---------------------------------------------------------------------------
__global__ void scatter_add_kernel(const float* __restrict__ feat,
                                   const int*   __restrict__ ei,
                                   float*       __restrict__ agg,
                                   int e) {
    int tid  = blockIdx.x * blockDim.x + threadIdx.x;
    int edge = tid >> 5;
    int lane = threadIdx.x & 31;
    if (edge >= e) return;
    int src = ei[edge];
    int dst = ei[NEDGES + edge];
    const float4 f = *(const float4*)(feat + (long)src * DH + lane * 4);
    float* ap = agg + (long)dst * DH + lane * 4;
    atomicAdd(ap + 0, f.x);
    atomicAdd(ap + 1, f.y);
    atomicAdd(ap + 2, f.z);
    atomicAdd(ap + 3, f.w);
}

// ---------------------------------------------------------------------------
// Fused SAGE GEMM:  out = act( (agg .* inv_deg) @ Wl^T + bl + xin @ Wr^T )
//
// One wave32 per 16-node row tile, 4 waves/block. K = 128 = 4 chunks of 32.
// Weights staged to LDS per block via global_load_async_to_lds_b128.
//
// A layout (16-bit 16x32, contiguous K, same as B): lane m = lane&15; element j of v16h:
//   j 0..7  -> K = kchunk*32 +      16*(lane>>4) + j
//   j 8..15 -> K = kchunk*32 + 16*(lane>>4) + j
// B layout (32x16): lane o = lane&15; element j -> K = kchunk*32 + 16*(lane>>4) + j
//   => a single contiguous 32B v16h read from row-major W[o][k] in LDS.
// C/D layout: VGPR r, lane l -> M = r + 8*(l>>4), N = l&15.
//
// ACT: 0 = none, 1 = relu, 2 = elu. OUT: 128 or 8 (8 => B padded with zeros).
// ---------------------------------------------------------------------------
template <int OUT, int ACT>
__global__ void sage_gemm_kernel(const float*    __restrict__ aggr,
                                 const float*    __restrict__ xin,
                                 const float*    __restrict__ deg_inv,
                                 const _Float16* __restrict__ Wl,
                                 const float*    __restrict__ bl,
                                 const _Float16* __restrict__ Wr,
                                 float*          __restrict__ out,
                                 int ntiles) {
    __shared__ __align__(32) _Float16 lwl[OUT * DH];
    __shared__ __align__(32) _Float16 lwr[OUT * DH];

    // ---- cooperative async stage of Wl, Wr into LDS (ASYNCcnt path) ----
    {
        const int      bytes  = OUT * DH * 2;
        const unsigned lbl    = (unsigned)(unsigned long)&lwl[0];
        const unsigned lbr    = (unsigned)(unsigned long)&lwr[0];
        const unsigned long g_l = (unsigned long)(const char*)Wl;
        const unsigned long g_r = (unsigned long)(const char*)Wr;
        for (int bo = threadIdx.x * 16; bo < bytes; bo += blockDim.x * 16) {
            asm volatile("global_load_async_to_lds_b128 %0, %1, off"
                         :: "v"(lbl + (unsigned)bo), "v"(g_l + bo) : "memory");
            asm volatile("global_load_async_to_lds_b128 %0, %1, off"
                         :: "v"(lbr + (unsigned)bo), "v"(g_r + bo) : "memory");
        }
        asm volatile("s_wait_asynccnt 0" ::: "memory");
    }
    __syncthreads();

    const int wave = threadIdx.x >> 5;
    const int lane = threadIdx.x & 31;
    const int tile = blockIdx.x * (blockDim.x >> 5) + wave;
    if (tile >= ntiles) return;                 // uniform per-wave exit: EXEC all-1s inside

    const int m0    = tile * 16;
    const int mrow  = lane & 15;
    const int khalf = lane >> 4;                // 0 or 1
    const int m     = m0 + mrow;
    const float inv = deg_inv[m];

    // ---- load + convert A tiles (agg*inv and x), K = 128 ----
    v16h a_agg[4], a_x[4];
    const float* arow = aggr + (long)m * DH;
    const float* xrow = xin  + (long)m * DH;
#pragma unroll
    for (int c = 0; c < 4; ++c) {
        const int kb0 = c * 32 + khalf * 16;      // elements 0..7
        const int kb1 = c * 32 + khalf * 16 + 8;  // elements 8..15
#pragma unroll
        for (int j = 0; j < 8; ++j) {
            a_agg[c][j]     = (_Float16)(arow[kb0 + j] * inv);
            a_agg[c][8 + j] = (_Float16)(arow[kb1 + j] * inv);
            a_x[c][j]       = (_Float16)(xrow[kb0 + j]);
            a_x[c][8 + j]   = (_Float16)(xrow[kb1 + j]);
        }
    }

    const int  ocol   = lane & 15;
    const int  obk    = lane >> 4;              // which K-16 group within chunk
    const int  nblk   = OUT / 16 > 0 ? OUT / 16 : 1;   // 8 for OUT=128, 1 for OUT=8
    const bool ovalid = (OUT >= 16) || (ocol < OUT);

#pragma unroll
    for (int ob = 0; ob < nblk; ++ob) {
        const int o = ob * 16 + ocol;

        // bias-initialized accumulator
        v8f acc;
        const float bv = ovalid ? bl[o] : 0.0f;
#pragma unroll
        for (int r = 0; r < 8; ++r) acc[r] = bv;

#pragma unroll
        for (int c = 0; c < 4; ++c) {
            v16h bwl, bwr;
            if (ovalid) {
                bwl = *(const v16h*)(&lwl[0] + o * DH + c * 32 + obk * 16);  // ds_load
                bwr = *(const v16h*)(&lwr[0] + o * DH + c * 32 + obk * 16);  // ds_load
            } else {
#pragma unroll
                for (int j = 0; j < 16; ++j) { bwl[j] = (_Float16)0.0f; bwr[j] = (_Float16)0.0f; }
            }
            acc = __builtin_amdgcn_wmma_f32_16x16x32_f16(
                false, a_agg[c], false, bwl, (short)0, acc, false, false);
            acc = __builtin_amdgcn_wmma_f32_16x16x32_f16(
                false, a_x[c],   false, bwr, (short)0, acc, false, false);
        }

        // activation + store (divergence only here, after all WMMAs of this block)
        if (ovalid) {
#pragma unroll
            for (int r = 0; r < 8; ++r) {
                float v = acc[r];
                if (ACT == 1) v = v > 0.0f ? v : 0.0f;
                else if (ACT == 2) v = v > 0.0f ? v : expm1f(v);
                const int mm = m0 + r + 8 * khalf;
                out[(long)mm * OUT + o] = v;
            }
        }
    }
}

// ---------------------------------------------------------------------------
// log_softmax over DOUT=8 columns, one thread per node
// ---------------------------------------------------------------------------
__global__ void log_softmax_kernel(const float* __restrict__ z, float* __restrict__ out, int n) {
    int i = blockIdx.x * blockDim.x + threadIdx.x;
    if (i >= n) return;
    float v[DOUT];
    float mx = -INFINITY;
#pragma unroll
    for (int j = 0; j < DOUT; ++j) { v[j] = z[(long)i * DOUT + j]; mx = fmaxf(mx, v[j]); }
    float s = 0.0f;
#pragma unroll
    for (int j = 0; j < DOUT; ++j) s += __expf(v[j] - mx);
    const float ls = __logf(s);
#pragma unroll
    for (int j = 0; j < DOUT; ++j) out[(long)i * DOUT + j] = v[j] - mx - ls;
}

// ---------------------------------------------------------------------------
// launch
// ---------------------------------------------------------------------------
static inline size_t align_up(size_t x, size_t a) { return (x + a - 1) & ~(a - 1); }

extern "C" void kernel_launch(void* const* d_in, const int* in_sizes, int n_in,
                              void* d_out, int out_size, void* d_ws, size_t ws_size,
                              hipStream_t stream) {
    (void)in_sizes; (void)n_in; (void)out_size; (void)ws_size;

    const float* x   = (const float*)d_in[0];
    const int*   ei  = (const int*)  d_in[1];   // [2, E] int32
    // d_in[2] = edge_attr (unused, matches reference)
    const float* W1l = (const float*)d_in[3];
    const float* b1l = (const float*)d_in[4];
    const float* W1r = (const float*)d_in[5];
    const float* Whl = (const float*)d_in[6];
    const float* bhl = (const float*)d_in[7];
    const float* Whr = (const float*)d_in[8];
    const float* W2l = (const float*)d_in[9];
    const float* b2l = (const float*)d_in[10];
    const float* W2r = (const float*)d_in[11];
    float* out = (float*)d_out;

    // ---- workspace layout (~79 MB) ----
    char* ws = (char*)d_ws;
    size_t off = 0;
    auto take = [&](size_t bytes) { char* p = ws + off; off = align_up(off + bytes, 256); return p; };

    float*    agg   = (float*)   take((size_t)NNODES * DH * 4);
    float*    h1    = (float*)   take((size_t)NNODES * DH * 4);
    float*    h2    = (float*)   take((size_t)NNODES * DH * 4);
    float*    z3    = (float*)   take((size_t)NNODES * DOUT * 4);
    int*      cnt   = (int*)     take((size_t)NNODES * 4);
    float*    dinv  = (float*)   take((size_t)NNODES * 4);
    _Float16* w1l16 = (_Float16*)take((size_t)DH * DIN * 2);
    _Float16* w1r16 = (_Float16*)take((size_t)DH * DIN * 2);
    _Float16* whl16 = (_Float16*)take((size_t)DH * DH * 2);
    _Float16* whr16 = (_Float16*)take((size_t)DH * DH * 2);
    _Float16* w2l16 = (_Float16*)take((size_t)DOUT * DH * 2);
    _Float16* w2r16 = (_Float16*)take((size_t)DOUT * DH * 2);

    const int ntiles    = NNODES / 16;                 // 3125
    const int gemm_blk  = 128;                         // 4 waves / block
    const int gemm_grid = (ntiles + 3) / 4;
    const int scat_grid = (NEDGES * 32 + 255) / 256;   // wave per edge

    // ---- degrees ----
    fill_i32<<<512, 256, 0, stream>>>(cnt, 0, NNODES);
    degree_kernel<<<(NEDGES + 255) / 256, 256, 0, stream>>>(ei, cnt, NEDGES);
    deg_inv_kernel<<<(NNODES + 255) / 256, 256, 0, stream>>>(cnt, dinv, NNODES);

    // ---- weight conversion (f32 -> f16) ----
    f32_to_f16<<<(DH * DIN + 255) / 256, 256, 0, stream>>>(W1l, w1l16, DH * DIN);
    f32_to_f16<<<(DH * DIN + 255) / 256, 256, 0, stream>>>(W1r, w1r16, DH * DIN);
    f32_to_f16<<<(DH * DH  + 255) / 256, 256, 0, stream>>>(Whl, whl16, DH * DH);
    f32_to_f16<<<(DH * DH  + 255) / 256, 256, 0, stream>>>(Whr, whr16, DH * DH);
    f32_to_f16<<<(DOUT * DH + 255) / 256, 256, 0, stream>>>(W2l, w2l16, DOUT * DH);
    f32_to_f16<<<(DOUT * DH + 255) / 256, 256, 0, stream>>>(W2r, w2r16, DOUT * DH);

    // ---- layer 1: relu(sage(x)) ----
    fill_f32<<<1024, 256, 0, stream>>>(agg, 0.0f, (long)NNODES * DH);
    scatter_add_kernel<<<scat_grid, 256, 0, stream>>>(x, ei, agg, NEDGES);
    sage_gemm_kernel<DH, 1><<<gemm_grid, gemm_blk, 0, stream>>>(
        agg, x, dinv, w1l16, b1l, w1r16, h1, ntiles);

    // ---- layer 2: elu(sage(h1)) ----
    fill_f32<<<1024, 256, 0, stream>>>(agg, 0.0f, (long)NNODES * DH);
    scatter_add_kernel<<<scat_grid, 256, 0, stream>>>(h1, ei, agg, NEDGES);
    sage_gemm_kernel<DH, 2><<<gemm_grid, gemm_blk, 0, stream>>>(
        agg, h1, dinv, whl16, bhl, whr16, h2, ntiles);

    // ---- layer 3: sage(h2) -> [N, 8] ----
    fill_f32<<<1024, 256, 0, stream>>>(agg, 0.0f, (long)NNODES * DH);
    scatter_add_kernel<<<scat_grid, 256, 0, stream>>>(h2, ei, agg, NEDGES);
    sage_gemm_kernel<DOUT, 0><<<gemm_grid, gemm_blk, 0, stream>>>(
        agg, h2, dinv, w2l16, b2l, w2r16, z3, ntiles);

    // ---- log_softmax ----
    log_softmax_kernel<<<(NNODES + 255) / 256, 256, 0, stream>>>(z3, out, NNODES);
}
